// MovingAverageGatedAttention_90984587199066
// MI455X (gfx1250) — compile-verified, matrix operand order checked by reading, not run on hardware
//
#include <hip/hip_runtime.h>
#include <hip/hip_bf16.h>

// ---------------- problem constants ----------------
constexpr int L_ = 2048, B_ = 8, E_ = 1024, Z_ = 128, H_ = 2048, MAXP_ = 2048;
constexpr int Mtot = L_ * B_;                 // 16384 rows in (L*B, .) layouts
constexpr int NP   = E_ + 2 * H_;             // 5120

typedef _Float16 half_t;
typedef __attribute__((ext_vector_type(16))) _Float16 v16h;
typedef __attribute__((ext_vector_type(8)))  _Float16 v8h;
typedef __attribute__((ext_vector_type(8)))  float    v8f;

__device__ __forceinline__ float sigf(float t) { return 1.0f / (1.0f + __expf(-t)); }
__device__ __forceinline__ float siluf(float t) { return t / (1.0f + __expf(-t)); }

// ==========================================================================
// f32 -> f16 conversion (weights)
// ==========================================================================
__global__ void f32_to_f16_kernel(const float* __restrict__ src,
                                  half_t* __restrict__ dst, long n) {
  long i = (long)blockIdx.x * blockDim.x + threadIdx.x;
  if (i < n) dst[i] = (half_t)src[i];
}

// ==========================================================================
// EMA: the FFT conv collapses to a 2-pole recurrence (kernel = sum of two
// decaying exponentials). One thread per (b,e) channel, coalesced across e.
// Also emits x in f16 for the Wp GEMM.
// ==========================================================================
__global__ void ema_kernel(const float* __restrict__ x,
                           const float* __restrict__ delta,
                           const float* __restrict__ alpha,
                           const float* __restrict__ ema_beta,
                           const float* __restrict__ ema_gamma,
                           const float* __restrict__ omega,
                           half_t* __restrict__ emah,
                           half_t* __restrict__ xh) {
  int t = blockIdx.x * blockDim.x + threadIdx.x;
  if (t >= B_ * E_) return;
  int b = t >> 10;          // / E_
  int e = t & (E_ - 1);
  const float scale = 0.70710678118654752f;   // sqrt(1/N), N=2
  float q[2], c[2], s[2] = {0.f, 0.f};
#pragma unroll
  for (int n = 0; n < 2; ++n) {
    float p = sigf(delta[e * 2 + n]);
    q[n] = 1.0f - p * sigf(alpha[e * 2 + n]);
    c[n] = p * ema_beta[e * 2 + n] * ema_gamma[e * 2 + n] * scale;
  }
  float om = omega[e];
  for (int l = 0; l < L_; ++l) {
    long idx = ((long)l * B_ + b) * E_ + e;
    float xv = x[idx];
    s[0] = q[0] * s[0] + xv;
    s[1] = q[1] * s[1] + xv;
    float y = c[0] * s[0] + c[1] * s[1] + xv * om;
    emah[idx] = (half_t)siluf(y);
    xh[idx]   = (half_t)xv;
  }
}

// ==========================================================================
// WMMA GEMM core: C[M,N] = A[M,K] (row-major f16) @ W[N,K]^T (row-major f16)
// 128 threads = 4 waves; block tile 128(M) x 64(N); each wave owns 32x64
// (2 M-frags x 4 N-frags = 8 accumulators) so 8 WMMAs amortize each set of
// LDS B-fragment loads. W tile double-buffered in LDS: one barrier per
// k-iteration; next tile staged while current is consumed.
// ==========================================================================
__device__ __forceinline__ void gemm_phase(const half_t* __restrict__ A,
                                           const half_t* __restrict__ W,
                                           int K, int m0, int n0,
                                           int lane, int tid,
                                           half_t* lds /* 2 x 64*32 halves */,
                                           v8f (&c)[2][4]) {
  const int arow = lane & 15;                  // fragment row / col-in-tile
  const int kh   = lane >> 4;                  // K-half select (ISA A layout)
  const int wn   = tid >> 1;                   // staging: W row within tile
  const int wk   = (tid & 1) * 16;             // staging: K chunk

  // prologue: stage k0 = 0 into buffer 0
  {
    const half_t* wp = W + (long)(n0 + wn) * K + wk;
    *(v8h*)(&lds[wn * 32 + wk])     = *(const v8h*)(wp);
    *(v8h*)(&lds[wn * 32 + wk + 8]) = *(const v8h*)(wp + 8);
  }
  __syncthreads();

  for (int k0 = 0; k0 < K; k0 += 32) {
    half_t* cur = lds + ((k0 >> 5) & 1) * (64 * 32);
    half_t* nxt = lds + (((k0 >> 5) + 1) & 1) * (64 * 32);

    // stage next W tile while current is consumed
    if (k0 + 32 < K) {
      const half_t* wp = W + (long)(n0 + wn) * K + (k0 + 32) + wk;
      *(v8h*)(&nxt[wn * 32 + wk])     = *(const v8h*)(wp);
      *(v8h*)(&nxt[wn * 32 + wk + 8]) = *(const v8h*)(wp + 8);
    }

    // two A fragments (16x32 each) straight from global
    v16h af[2];
#pragma unroll
    for (int mt = 0; mt < 2; ++mt) {
      const half_t* ap = A + (long)(m0 + mt * 16 + arow) * K + k0 + kh * 8;
      __builtin_prefetch(ap + 32, 0, 1);       // global_prefetch_b8 next block
      v8h a0 = *(const v8h*)(ap);
      v8h a1 = *(const v8h*)(ap + 16);
#pragma unroll
      for (int i = 0; i < 8; ++i) { af[mt][i] = a0[i]; af[mt][8 + i] = a1[i]; }
    }

    // 4 N-subtiles from LDS; 8 WMMAs total per iteration
#pragma unroll
    for (int nt = 0; nt < 4; ++nt) {
      const half_t* bpt = &cur[(nt * 16 + arow) * 32 + kh * 8];
      v8h b0 = *(const v8h*)(bpt);
      v8h b1 = *(const v8h*)(bpt + 16);
      v16h bf;
#pragma unroll
      for (int i = 0; i < 8; ++i) { bf[i] = b0[i]; bf[8 + i] = b1[i]; }
#pragma unroll
      for (int mt = 0; mt < 2; ++mt) {
        c[mt][nt] = __builtin_amdgcn_wmma_f32_16x16x32_f16(
            /*neg_a=*/false, af[mt], /*neg_b=*/false, bf,
            /*c_mod=*/(short)0, c[mt][nt], /*reuse_a=*/false, /*reuse_b=*/false);
      }
    }
    __syncthreads();
  }
}

template <class Epi>
__device__ __forceinline__ void gemm_epilogue(int bat, int m0, int n0, int lane,
                                              const v8f (&c)[2][4], const Epi& epi) {
  const int mhi  = (lane < 16) ? 0 : 8;        // CDNA5 16x16 f32 C/D layout
  const int ncol = lane & 15;
#pragma unroll
  for (int mt = 0; mt < 2; ++mt)
#pragma unroll
    for (int nt = 0; nt < 4; ++nt)
#pragma unroll
      for (int v = 0; v < 8; ++v)
        epi(bat, m0 + mt * 16 + mhi + v, n0 + nt * 16 + ncol, c[mt][nt][v]);
}

template <class Epi>
__global__ __launch_bounds__(128)
void gemm_wmma_kernel(const half_t* __restrict__ Abase,
                      const half_t* __restrict__ Wbase,
                      int K, long strideAbatch, long strideWbatch, Epi epi) {
  __shared__ __align__(16) half_t lds[2 * 64 * 32];
  const int tid = threadIdx.x, lane = tid & 31, wave = tid >> 5;
  const int bat = blockIdx.z;
  const half_t* A = Abase + (long)bat * strideAbatch;
  const half_t* W = Wbase + (long)bat * strideWbatch;
  const int m0 = blockIdx.x * 128 + wave * 32;
  const int n0 = blockIdx.y * 64;
  v8f c[2][4] = {};
  gemm_phase(A, W, K, m0, n0, lane, tid, lds, c);
  gemm_epilogue(bat, m0, n0, lane, c, epi);
}

// Fused dual-GEMM: C = A1@W1^T + A2@W2^T (same output tile, two K loops with
// persistent accumulators) + epilogue. Removes the 64MB f32 acc buffer and a
// full extra pass (saves ~192MB of HBM traffic).
template <class Epi>
__global__ __launch_bounds__(128)
void gemm2_wmma_kernel(const half_t* __restrict__ A1, const half_t* __restrict__ W1, int K1,
                       const half_t* __restrict__ A2, const half_t* __restrict__ W2, int K2,
                       Epi epi) {
  __shared__ __align__(16) half_t lds[2 * 64 * 32];
  const int tid = threadIdx.x, lane = tid & 31, wave = tid >> 5;
  const int m0 = blockIdx.x * 128 + wave * 32;
  const int n0 = blockIdx.y * 64;
  v8f c[2][4] = {};
  gemm_phase(A1, W1, K1, m0, n0, lane, tid, lds, c);
  gemm_phase(A2, W2, K2, m0, n0, lane, tid, lds, c);
  gemm_epilogue(0, m0, n0, lane, c, epi);
}

// ==========================================================================
// Epilogues
// ==========================================================================
struct EpiMx {  // mx = emaout @ Wm^T + bm  -> f16
  half_t* mx; const float* bm;
  __device__ void operator()(int, int m, int n, float v) const {
    mx[(long)m * Z_ + n] = (half_t)(v + bm[n]);
  }
};

struct EpiZ {   // z = silu(mx@Wz^T+bz); qr/kr = z*gamma+beta, (B,L,Z) layout
  half_t* qr; half_t* kr; const float* bz; const float* gamma; const float* beta;
  __device__ void operator()(int, int m, int n, float v) const {
    float t = siluf(v + bz[n]);
    int l = m >> 3, b = m & 7;                 // m = l*B + b, B=8
    long o = ((long)b * L_ + l) * Z_ + n;
    qr[o] = (half_t)(t * gamma[n]      + beta[n]);
    kr[o] = (half_t)(t * gamma[Z_ + n] + beta[Z_ + n]);
  }
};

struct EpiBase { // base = x@Wp^T+bp; u=sigmoid, r=silu, v=silu stored TRANSPOSED
  half_t* u; half_t* r; half_t* vT; const float* bp;
  __device__ void operator()(int, int m, int n, float val) const {
    float t = val + bp[n];
    if (n < E_) {
      u[(long)m * E_ + n] = (half_t)sigf(t);
    } else {
      float s = siluf(t);
      if (n < E_ + H_) {
        r[(long)m * H_ + (n - E_)] = (half_t)s;
      } else {
        int l = m >> 3, b = m & 7;
        // vT[b][h][l] : (N=H, K=L) weight-format so the AV GEMM is standard
        vT[((long)b * H_ + (n - E_ - H_)) * L_ + l] = (half_t)s;
      }
    }
  }
};

struct EpiAttn { // S = relu(qk/L + relbias)^2, causal (no softmax in MEGA)
  half_t* S; const float* rpw;
  __device__ void operator()(int b, int l, int mc, float v) const {
    float val = v * (1.0f / (float)L_) + rpw[MAXP_ - 1 + mc - l];
    val = (mc <= l) ? fmaxf(val, 0.0f) : 0.0f;
    val *= val;
    S[((long)b * L_ + l) * L_ + mc] = (half_t)val;
  }
};

struct EpiAV {  // h = S @ v ; fused hr = h * r  in (L*B, H) layout
  half_t* hr; const half_t* r;
  __device__ void operator()(int b, int l, int h, float v) const {
    long o = ((long)l * B_ + b) * H_ + h;
    hr[o] = (half_t)(v * (float)r[o]);
  }
};

struct EpiFinal { // h2 = tanh(acc + bhu + bhw); out = x + u*(h2 - x)
  const float* x; const half_t* u; const float* bhu; const float* bhw; float* out;
  __device__ void operator()(int, int m, int n, float v) const {
    long i = (long)m * E_ + n;
    float h  = tanhf(v + bhu[n] + bhw[n]);
    float xv = x[i];
    out[i] = xv + (float)u[i] * (h - xv);
  }
};

// ==========================================================================
// Host launcher
// ==========================================================================
extern "C" void kernel_launch(void* const* d_in, const int* in_sizes, int n_in,
                              void* d_out, int out_size, void* d_ws, size_t ws_size,
                              hipStream_t stream) {
  (void)in_sizes; (void)n_in; (void)out_size; (void)ws_size;

  const float* x     = (const float*)d_in[0];
  /* d_in[1] = attn_mask (causality applied analytically) */
  const float* delta = (const float*)d_in[2];
  const float* alpha = (const float*)d_in[3];
  const float* emab  = (const float*)d_in[4];
  const float* emag  = (const float*)d_in[5];
  const float* omega = (const float*)d_in[6];
  const float* Wm    = (const float*)d_in[7];
  const float* bm    = (const float*)d_in[8];
  const float* Wz    = (const float*)d_in[9];
  const float* bz    = (const float*)d_in[10];
  const float* Wp    = (const float*)d_in[11];
  const float* bp    = (const float*)d_in[12];
  const float* Whw   = (const float*)d_in[13];
  const float* bhw   = (const float*)d_in[14];
  const float* Whu   = (const float*)d_in[15];
  const float* bhu   = (const float*)d_in[16];
  const float* gamma = (const float*)d_in[17];
  const float* beta  = (const float*)d_in[18];
  const float* rpw   = (const float*)d_in[19];
  float* out = (float*)d_out;

  // ---- workspace carve-up ----
  char* wsp = (char*)d_ws;
  auto alloc = [&](size_t bytes) {
    char* p = wsp; wsp += (bytes + 255) & ~(size_t)255; return p;
  };
  half_t* xh    = (half_t*)alloc((size_t)Mtot * E_ * 2);
  half_t* emah  = (half_t*)alloc((size_t)Mtot * E_ * 2);
  half_t* mxh   = (half_t*)alloc((size_t)Mtot * Z_ * 2);
  half_t* qrh   = (half_t*)alloc((size_t)B_ * L_ * Z_ * 2);
  half_t* krh   = (half_t*)alloc((size_t)B_ * L_ * Z_ * 2);
  half_t* uh    = (half_t*)alloc((size_t)Mtot * E_ * 2);
  half_t* rh    = (half_t*)alloc((size_t)Mtot * H_ * 2);
  half_t* vTh   = (half_t*)alloc((size_t)B_ * H_ * L_ * 2);
  half_t* Sh    = (half_t*)alloc((size_t)B_ * L_ * L_ * 2);
  half_t* hrh   = (half_t*)alloc((size_t)Mtot * H_ * 2);
  half_t* Wmh   = (half_t*)alloc((size_t)Z_ * E_ * 2);
  half_t* Wzh   = (half_t*)alloc((size_t)Z_ * Z_ * 2);
  half_t* Wph   = (half_t*)alloc((size_t)NP * E_ * 2);
  half_t* Whwh  = (half_t*)alloc((size_t)E_ * H_ * 2);
  half_t* Whuh  = (half_t*)alloc((size_t)E_ * Z_ * 2);

  auto cvt = [&](const float* s, half_t* d, long n) {
    f32_to_f16_kernel<<<dim3((unsigned)((n + 255) / 256)), dim3(256), 0, stream>>>(s, d, n);
  };
  cvt(Wm,  Wmh,  (long)Z_ * E_);
  cvt(Wz,  Wzh,  (long)Z_ * Z_);
  cvt(Wp,  Wph,  (long)NP * E_);
  cvt(Whw, Whwh, (long)E_ * H_);
  cvt(Whu, Whuh, (long)E_ * Z_);

  // EMA recurrence + x->f16
  ema_kernel<<<dim3(B_ * E_ / 256), dim3(256), 0, stream>>>(
      x, delta, alpha, emab, emag, omega, emah, xh);

  // mx = emaout @ Wm^T + bm          [16384 x 128, K=1024]
  gemm_wmma_kernel<EpiMx><<<dim3(Mtot / 128, Z_ / 64, 1), dim3(128), 0, stream>>>(
      emah, Wmh, E_, 0, 0, EpiMx{mxh, bm});

  // z = silu(mx @ Wz^T + bz) -> qr/kr [16384 x 128, K=128]
  gemm_wmma_kernel<EpiZ><<<dim3(Mtot / 128, Z_ / 64, 1), dim3(128), 0, stream>>>(
      mxh, Wzh, Z_, 0, 0, EpiZ{qrh, krh, bz, gamma, beta});

  // base = x @ Wp^T + bp -> u, r, vT  [16384 x 5120, K=1024]
  gemm_wmma_kernel<EpiBase><<<dim3(Mtot / 128, NP / 64, 1), dim3(128), 0, stream>>>(
      xh, Wph, E_, 0, 0, EpiBase{uh, rh, vTh, bp});

  // S = relu(qr@kr^T/L + bias)^2, causal   [per batch: 2048x2048, K=128]
  gemm_wmma_kernel<EpiAttn><<<dim3(L_ / 128, L_ / 64, B_), dim3(128), 0, stream>>>(
      qrh, krh, Z_, (long)L_ * Z_, (long)L_ * Z_, EpiAttn{Sh, rpw});

  // h = S @ v ; hr = h * r                 [per batch: 2048x2048, K=2048]
  gemm_wmma_kernel<EpiAV><<<dim3(L_ / 128, H_ / 64, B_), dim3(128), 0, stream>>>(
      Sh, vTh, L_, (long)L_ * L_, (long)H_ * L_, EpiAV{hrh, rh});

  // fused: tanh(mx@Whu^T + hr@Whw^T + bhu + bhw), out = x + u*(h - x)
  //                                       [16384 x 1024, K=128 then K=2048]
  gemm2_wmma_kernel<EpiFinal><<<dim3(Mtot / 128, E_ / 64, 1), dim3(128), 0, stream>>>(
      mxh, Whuh, Z_, hrh, Whwh, H_, EpiFinal{x, uh, bhu, bhw, out});
}
